// FuseBlock_20435454395197
// MI455X (gfx1250) — compile-verified, matrix operand order checked by reading, not run on hardware
//
#include <hip/hip_runtime.h>
#include <hip/hip_bf16.h>

// ---------------------------------------------------------------------------
// FuseBlock for MI455X (gfx1250): all GEMMs on v_wmma_f32_16x16x32_bf16.
//  - projections fb/fc/fd: TDM-staged A-panel in LDS + 2-wide N register
//    blocking (tensor_load_to_lds / s_wait_tensorcnt / ds reads)
//  - flash spatial attention: double-buffered TDM pipeline for the V panel
//  - channel attention (att=fa*faT, softmax, att*fa)
//  - 3x3 conv as implicit GEMM: cooperative LDS im2col tile shared by 4
//    M-tile waves per block + WMMA; BN(train) + ReLU + y*sa*ca
// Requires ~164 MB of workspace (phase-aliased).
// ---------------------------------------------------------------------------

typedef __attribute__((ext_vector_type(16))) __bf16 v16bf;
typedef __attribute__((ext_vector_type(8)))  float  v8f;
typedef __attribute__((ext_vector_type(4)))  unsigned int v4u;
typedef __attribute__((ext_vector_type(4)))  int v4i;
typedef __attribute__((ext_vector_type(8)))  int v8i;

#define DEV static __device__ __forceinline__

// K index held by half h (0..15) of a 16-bit A/B fragment for lane L
// (CDNA5 ISA 7.12.2: lanes 0-15 get K {0..7,16..23}, lanes 16-31 get +8)
DEV int kmap(int h, int lane) {
  return (h < 8 ? h : h + 8) + ((lane & 16) ? 8 : 0);
}

DEV unsigned short f2bf(float f) {           // RNE float -> bf16 bits
  unsigned int u = __float_as_uint(f);
  u += 0x7FFFu + ((u >> 16) & 1u);
  return (unsigned short)(u >> 16);
}

union Frag16 { unsigned short u[16]; v16bf v; };

DEV v16bf load_frag(const unsigned short* __restrict__ p) {
  return *reinterpret_cast<const v16bf*>(p);
}

DEV v8f wmma_bf16(v16bf a, v16bf b, v8f c) {
  return __builtin_amdgcn_wmma_f32_16x16x32_bf16(false, a, false, b,
                                                 (short)0, c, false, false);
}

// ---------------------------------------------------------------------------
// Tensor Data Mover: 2-D tile copy Global -> LDS.
// D# per CDNA5 ISA ch.8: group0 {count=1, lds_addr, global_addr, type=2},
// group1 {data_size=4B, tensor_dim0/1, tile_dim0/1, dim0 stride}.
// rows x row_dwords dwords, row pitch = stride_dwords.
// ---------------------------------------------------------------------------
DEV void tdm_load_2d(void* lds_dst, const void* gsrc,
                     unsigned row_dwords, unsigned rows,
                     unsigned long long stride_dwords) {
  unsigned lds = (unsigned)(unsigned long long)lds_dst;   // low 32b = LDS offset
  unsigned long long ga = (unsigned long long)gsrc;
  v4u g0;
  g0[0] = 1u;                                             // count=1 (user D#)
  g0[1] = lds;                                            // lds_addr
  g0[2] = (unsigned)(ga & 0xffffffffu);                   // global_addr[31:0]
  g0[3] = (unsigned)((ga >> 32) & 0x01ffffffu) | (2u << 30); // ga[56:32]|type=2
  v8i g1;
  g1[0] = (int)(2u << 16);                                // data_size=4B
  g1[1] = (int)((row_dwords & 0xffffu) << 16);            // tensor_dim0 lo
  g1[2] = (int)(((row_dwords >> 16) & 0xffffu) |          // tensor_dim0 hi
                ((rows & 0xffffu) << 16));                // tensor_dim1 lo
  g1[3] = (int)(((rows >> 16) & 0xffffu) |                // tensor_dim1 hi
                ((row_dwords & 0xffffu) << 16));          // tile_dim0
  g1[4] = (int)(rows & 0xffffu);                          // tile_dim1 (tile_dim2=0)
  g1[5] = (int)(stride_dwords & 0xffffffffu);             // dim0 stride lo
  g1[6] = (int)((stride_dwords >> 32) & 0xffffu);         // dim0 stride hi
  g1[7] = 0;
  v4i z4 = {0, 0, 0, 0};
#if __has_include(<hip/amd_detail/amd_gfx1250_TDM.h>)
  v8i z8 = {0, 0, 0, 0, 0, 0, 0, 0};                      // clang-23 arity
  __builtin_amdgcn_tensor_load_to_lds(g0, g1, z4, z4, z8, 0);
#else
  __builtin_amdgcn_tensor_load_to_lds(g0, g1, z4, z4, 0); // ROCm 7.2 arity
#endif
}

DEV void tdm_wait() {
  __builtin_amdgcn_s_wait_tensorcnt(0);
  asm volatile("" ::: "memory");
}

// ---------------------------------------------------------------------------
// Pack f32 matrix element (m,k) = src[m*rs + k*cs] into A-fragment layout:
// dst tile (mt*KT+kt), per lane 16 contiguous bf16 halfs.
// ---------------------------------------------------------------------------
__global__ void pack_a_kernel(const float* __restrict__ src,
                              unsigned short* __restrict__ dst,
                              int MT, int KT, long rs, long cs,
                              long sBatch, long dBatch) {
  long total = (long)MT * KT * 512;
  long gid = (long)blockIdx.x * blockDim.x + threadIdx.x;
  if (gid >= total) return;
  int b = blockIdx.y;
  long tile = gid >> 9;
  int within = (int)(gid & 511);
  int lane = within >> 4, h = within & 15;
  int mt = (int)(tile / KT), kt = (int)(tile % KT);
  int m = mt * 16 + (lane & 15);
  int k = kt * 32 + kmap(h, lane);
  dst[(long)b * dBatch + gid] = f2bf(src[(long)b * sBatch + (long)m * rs + (long)k * cs]);
}

// B-fragment layout: element (k,n) = src[k*rs + n*cs]; tile (nt*KT+kt).
__global__ void pack_b_kernel(const float* __restrict__ src,
                              unsigned short* __restrict__ dst,
                              int NT, int KT, long rs, long cs,
                              long sBatch, long dBatch) {
  long total = (long)NT * KT * 512;
  long gid = (long)blockIdx.x * blockDim.x + threadIdx.x;
  if (gid >= total) return;
  int b = blockIdx.y;
  long tile = gid >> 9;
  int within = (int)(gid & 511);
  int lane = within >> 4, h = within & 15;
  int nt = (int)(tile / KT), kt = (int)(tile % KT);
  int n = nt * 16 + (lane & 15);
  int k = kt * 32 + kmap(h, lane);
  dst[(long)b * dBatch + gid] = f2bf(src[(long)b * sBatch + (long)k * rs + (long)n * cs]);
}

// common GEMM epilogue: out[b*oBatch + m*ldOut + n] = s*acc + bias[m] + res
DEV void gemm_epilogue(v8f acc, int mt, int nt, int b, int lane, long ldOut,
                       const float* __restrict__ bias, float s,
                       const float* __restrict__ residual, long oBatch,
                       float* __restrict__ out) {
  int n  = nt * 16 + (lane & 15);
  int hi = (lane & 16) ? 8 : 0;
#pragma unroll
  for (int r = 0; r < 8; ++r) {
    int m = mt * 16 + r + hi;
    long idx = (long)b * oBatch + (long)m * ldOut + n;
    float v = acc[r] * s;
    if (bias)     v += bias[m];
    if (residual) v += residual[idx];
    out[idx] = v;
  }
}

// ---------------------------------------------------------------------------
// Plain WMMA GEMM (used for K=4096 channel-attention GEMM): 4 waves/block,
// one 16x16 tile per wave, operands streamed from global.
// ---------------------------------------------------------------------------
__global__ void gemm_bf16_kernel(const unsigned short* __restrict__ Ap,
                                 const unsigned short* __restrict__ Bp,
                                 float* __restrict__ out,
                                 int NT, int KT, long ldOut,
                                 const float* __restrict__ bias,
                                 const float* __restrict__ scalePtr,
                                 const float* __restrict__ residual,
                                 long aBatch, long bBatch, long oBatch) {
  int lane = threadIdx.x & 31;
  int wid  = threadIdx.x >> 5;
  int nt = blockIdx.x * 4 + wid;
  if (nt >= NT) return;
  int mt = blockIdx.y;
  int b  = blockIdx.z;
  const unsigned short* Abase = Ap + (long)b * aBatch + ((long)mt * KT) * 512 + lane * 16;
  const unsigned short* Bbase = Bp + (long)b * bBatch + ((long)nt * KT) * 512 + lane * 16;
  v8f acc = {};
  for (int kt = 0; kt < KT; ++kt) {
    if (kt + 1 < KT) {
      __builtin_prefetch(Abase + (kt + 1) * 512, 0, 3);
      __builtin_prefetch(Bbase + (kt + 1) * 512, 0, 3);
    }
    v16bf a  = load_frag(Abase + kt * 512);
    v16bf bf = load_frag(Bbase + kt * 512);
    acc = wmma_bf16(a, bf, acc);
  }
  float s = scalePtr ? scalePtr[0] : 1.0f;
  gemm_epilogue(acc, mt, nt, b, lane, ldOut, bias, s, residual, oBatch, out);
}

// ---------------------------------------------------------------------------
// TDM-staged GEMM: the A K-panel (KT tiles, contiguous) is DMA'd once per
// block into LDS by the Tensor Data Mover and shared by all 4 waves; each
// wave computes two N-tiles, reusing the A fragment across two WMMAs.
// ---------------------------------------------------------------------------
template <int KT>
__global__ void gemm_lds_kernel(const unsigned short* __restrict__ Ap,
                                const unsigned short* __restrict__ Bp,
                                float* __restrict__ out,
                                int NT, long ldOut,
                                const float* __restrict__ bias,
                                const float* __restrict__ scalePtr,
                                const float* __restrict__ residual,
                                long aBatch, long bBatch, long oBatch) {
  __shared__ unsigned short As[KT * 512];
  int lane = threadIdx.x & 31;
  int wid  = threadIdx.x >> 5;
  int mt = blockIdx.y;
  int b  = blockIdx.z;
  if (wid == 0) {
    const unsigned short* Aglob = Ap + (long)b * aBatch + ((long)mt * KT) * 512;
    tdm_load_2d(As, Aglob, KT * 256u, 1u, KT * 256ull);   // 1 row of KT KB
    tdm_wait();
  }
  __syncthreads();

  int nt0 = (blockIdx.x * 4 + wid) * 2;
  if (nt0 < NT) {
    const unsigned short* B0 = Bp + (long)b * bBatch + ((long)nt0 * KT) * 512 + lane * 16;
    const unsigned short* B1 = B0 + KT * 512;
    v8f acc0 = {}, acc1 = {};
#pragma unroll
    for (int kt = 0; kt < KT; ++kt) {
      v16bf a  = load_frag(As + kt * 512 + lane * 16);    // LDS
      v16bf b0 = load_frag(B0 + kt * 512);
      v16bf b1 = load_frag(B1 + kt * 512);
      acc0 = wmma_bf16(a, b0, acc0);
      acc1 = wmma_bf16(a, b1, acc1);
    }
    float s = scalePtr ? scalePtr[0] : 1.0f;
    gemm_epilogue(acc0, mt, nt0,     b, lane, ldOut, bias, s, residual, oBatch, out);
    gemm_epilogue(acc1, mt, nt0 + 1, b, lane, ldOut, bias, s, residual, oBatch, out);
  }
}

// ---------------------------------------------------------------------------
// Flash spatial attention: one wave per 16-query tile per batch.
// Q = fb^T (4096x32), K = fc^T, V = fd^T (4096x256); seq 4096.
// V chunk panels (16 tiles x 1KB, 2MB row pitch) are double-buffered into
// LDS by the TDM while the wave runs S/softmax/PV WMMAs on the other buffer.
// sa = alpha * O + x_low  written directly.
// ---------------------------------------------------------------------------
__global__ void attn_kernel(const unsigned short* __restrict__ Qp,
                            const unsigned short* __restrict__ Kp,
                            const unsigned short* __restrict__ Vp,
                            const float* __restrict__ x_low,
                            const float* __restrict__ alphaPtr,
                            float* __restrict__ sa) {
  __shared__ unsigned short Plds[16][36];
  __shared__ unsigned short Vbuf[2][16 * 512];
  int lane = threadIdx.x;
  int qt = blockIdx.x;         // query tile 0..255
  int b  = blockIdx.y;
  int hi = (lane & 16) ? 8 : 0;

  v16bf q = load_frag(Qp + ((long)(b * 256 + qt)) * 512 + lane * 16);
  v8f o[16];
#pragma unroll
  for (int f = 0; f < 16; ++f) o[f] = (v8f){};
  float mrow[8], lsum[8];
#pragma unroll
  for (int r = 0; r < 8; ++r) { mrow[r] = -3.0e38f; lsum[r] = 0.f; }

  // prologue: stage V panel for chunk 0 (16 rows x 256 dwords, pitch 512K dwords)
  tdm_load_2d(Vbuf[0], Vp + ((long)(b * 128)) * 512, 256u, 16u, 524288ull);
  tdm_wait();

  for (int ch = 0; ch < 128; ++ch) {           // 32 keys per chunk
    int cur = ch & 1;
    if (ch + 1 < 128)                          // async: next V panel
      tdm_load_2d(Vbuf[1 - cur], Vp + ((long)(b * 128 + ch + 1)) * 512,
                  256u, 16u, 524288ull);

    v16bf k0 = load_frag(Kp + ((long)(b * 256 + ch * 2 + 0)) * 512 + lane * 16);
    v16bf k1 = load_frag(Kp + ((long)(b * 256 + ch * 2 + 1)) * 512 + lane * 16);
    v8f z = {};
    v8f s0 = wmma_bf16(q, k0, z);
    v8f s1 = wmma_bf16(q, k1, z);

    float p0[8], p1[8], corr[8];
#pragma unroll
    for (int r = 0; r < 8; ++r) {
      float c = fmaxf(s0[r], s1[r]);
      c = fmaxf(c, __shfl_xor(c, 1, 32));
      c = fmaxf(c, __shfl_xor(c, 2, 32));
      c = fmaxf(c, __shfl_xor(c, 4, 32));
      c = fmaxf(c, __shfl_xor(c, 8, 32));      // masks <16 keep row halves
      float mn = fmaxf(mrow[r], c);
      corr[r] = __expf(mrow[r] - mn);
      p0[r] = __expf(s0[r] - mn);
      p1[r] = __expf(s1[r] - mn);
      float su = p0[r] + p1[r];
      su += __shfl_xor(su, 1, 32);
      su += __shfl_xor(su, 2, 32);
      su += __shfl_xor(su, 4, 32);
      su += __shfl_xor(su, 8, 32);
      lsum[r] = lsum[r] * corr[r] + su;
      mrow[r] = mn;
    }
#pragma unroll
    for (int f = 0; f < 16; ++f)
#pragma unroll
      for (int r = 0; r < 8; ++r) o[f][r] *= corr[r];

    // transpose P (C-frag layout) -> A-frag layout via LDS
#pragma unroll
    for (int r = 0; r < 8; ++r) {
      Plds[r + hi][(lane & 15)]      = f2bf(p0[r]);
      Plds[r + hi][16 + (lane & 15)] = f2bf(p1[r]);
    }
    __syncthreads();
    Frag16 pf;
#pragma unroll
    for (int h = 0; h < 16; ++h) pf.u[h] = Plds[lane & 15][kmap(h, lane)];
    v16bf pA = pf.v;

#pragma unroll
    for (int f = 0; f < 16; ++f) {
      v16bf vb = load_frag(Vbuf[cur] + f * 512 + lane * 16);   // LDS (TDM-staged)
      o[f] = wmma_bf16(pA, vb, o[f]);
    }
    __syncthreads();
    if (ch + 1 < 128) tdm_wait();              // next buffer ready
  }

  float alpha = alphaPtr[0];
#pragma unroll
  for (int f = 0; f < 16; ++f)
#pragma unroll
    for (int r = 0; r < 8; ++r) {
      float val = o[f][r] / lsum[r];
      int och = f * 16 + (lane & 15);
      int n   = qt * 16 + r + hi;
      long idx = ((long)(b * 256 + och)) * 4096 + n;
      sa[idx] = alpha * val + x_low[idx];
    }
}

// ---------------------------------------------------------------------------
// Channel-attention softmax: softmax over d of (rowmax - att) == exp(rowmin-a)/sum
// one 256-thread block per (b,c) row.
// ---------------------------------------------------------------------------
__global__ void ca_softmax_kernel(const float* __restrict__ att,
                                  float* __restrict__ outp) {
  __shared__ float red[256];
  long row = blockIdx.x;
  int t = threadIdx.x;
  float a = att[row * 256 + t];
  red[t] = a; __syncthreads();
  for (int s = 128; s > 0; s >>= 1) { if (t < s) red[t] = fminf(red[t], red[t + s]); __syncthreads(); }
  float mn = red[0]; __syncthreads();
  float e = __expf(mn - a);
  red[t] = e; __syncthreads();
  for (int s = 128; s > 0; s >>= 1) { if (t < s) red[t] += red[t + s]; __syncthreads(); }
  outp[row * 256 + t] = e / red[0];
}

__global__ void add_kernel(const float* __restrict__ a, const float* __restrict__ b,
                           float* __restrict__ c) {
  long i = (long)blockIdx.x * 256 + threadIdx.x;
  c[i] = a[i] + b[i];
}

// ---------------------------------------------------------------------------
// 3x3 conv as implicit GEMM: M=256(o), K=2304 (k = c*9 + tap), N = b*h*w.
// Block = 128 threads (4 waves) covering ONE pixel tile and FOUR M-tiles:
// the 32x16 im2col B-tile is built cooperatively in LDS once per k-step
// (coalesced gathers) and shared by all 4 waves' WMMAs.  y -> d_out.
// ---------------------------------------------------------------------------
__global__ void conv_kernel(const unsigned short* __restrict__ WA,
                            const float* __restrict__ xsum,
                            float* __restrict__ y) {
  __shared__ unsigned short Btile[32][20];   // 32 k-rows x 16 cols (+pad)
  int tid  = threadIdx.x;                    // 128
  int lane = tid & 31, wid = tid >> 5;
  int nt = blockIdx.x;                       // pixel tile 0..2047
  int mt = blockIdx.y * 4 + wid;             // M tile 0..15
  int b = nt >> 8, rem = nt & 255;
  int h = rem >> 2, w0 = (rem & 3) * 16;
  v8f acc = {};
  for (int kt = 0; kt < 72; ++kt) {
    __syncthreads();                         // previous k-step reads done
#pragma unroll
    for (int e = 0; e < 4; ++e) {            // 512 elems / 128 threads
      int idx = e * 128 + tid;               // consecutive tid -> consecutive wx
      int kk = idx >> 4, n = idx & 15;
      int k = kt * 32 + kk;
      int c = k / 9, tap = k % 9;
      int hy = h + tap / 3 - 1, wx = w0 + n + tap % 3 - 1;
      float v = 0.f;
      if ((unsigned)hy < 64u && (unsigned)wx < 64u)
        v = xsum[((long)(b * 256 + c)) * 4096 + hy * 64 + wx];
      Btile[kk][n] = f2bf(v);
    }
    __syncthreads();
    Frag16 fr;
#pragma unroll
    for (int hh = 0; hh < 16; ++hh)
      fr.u[hh] = Btile[kmap(hh, lane)][lane & 15];
    v16bf a = load_frag(WA + ((long)mt * 72 + kt) * 512 + lane * 16);
    acc = wmma_bf16(a, fr.v, acc);
  }
  int hi = (lane & 16) ? 8 : 0;
#pragma unroll
  for (int r = 0; r < 8; ++r) {
    int o = mt * 16 + r + hi;
    y[((long)(b * 256 + o)) * 4096 + h * 64 + w0 + (lane & 15)] = acc[r];
  }
}

// BN train-mode stats per channel over (N,H,W); emit scale/shift.
__global__ void bn_stats_kernel(const float* __restrict__ y,
                                const float* __restrict__ gamma,
                                const float* __restrict__ bnbeta,
                                float* __restrict__ scale,
                                float* __restrict__ shift) {
  __shared__ float s1[256], s2[256];
  int o = blockIdx.x, t = threadIdx.x;
  float a = 0.f, q = 0.f;
  for (int bb = 0; bb < 8; ++bb) {
    const float* p = y + ((long)(bb * 256 + o)) * 4096;
    for (int i = t; i < 4096; i += 256) { float v = p[i]; a += v; q += v * v; }
  }
  s1[t] = a; s2[t] = q; __syncthreads();
  for (int s = 128; s > 0; s >>= 1) {
    if (t < s) { s1[t] += s1[t + s]; s2[t] += s2[t + s]; }
    __syncthreads();
  }
  if (t == 0) {
    float N = 8.0f * 4096.0f;
    float mean = s1[0] / N;
    float var  = s2[0] / N - mean * mean;
    float inv  = rsqrtf(var + 1e-5f);
    float sc   = gamma[o] * inv;
    scale[o] = sc;
    shift[o] = bnbeta[o] - mean * sc;
  }
}

// out = relu(y*scale+shift) * sa * ca   (in place over d_out)
__global__ void final_kernel(float* __restrict__ y,
                             const float* __restrict__ sa,
                             const float* __restrict__ ca,
                             const float* __restrict__ scale,
                             const float* __restrict__ shift) {
  long i = (long)blockIdx.x * 256 + threadIdx.x;
  int chn = (int)((i >> 12) & 255);
  float v = fmaxf(y[i] * scale[chn] + shift[chn], 0.f);
  y[i] = v * sa[i] * ca[i];
}

// ---------------------------------------------------------------------------
extern "C" void kernel_launch(void* const* d_in, const int* in_sizes, int n_in,
                              void* d_out, int out_size, void* d_ws, size_t ws_size,
                              hipStream_t stream) {
  (void)in_sizes; (void)n_in; (void)out_size; (void)ws_size;
  const float* x_low  = (const float*)d_in[0];
  const float* x_high = (const float*)d_in[1];
  const float* wb = (const float*)d_in[2];
  const float* bb = (const float*)d_in[3];
  const float* wc = (const float*)d_in[4];
  const float* bc = (const float*)d_in[5];
  const float* wd = (const float*)d_in[6];
  const float* bd = (const float*)d_in[7];
  const float* alpha = (const float*)d_in[8];
  const float* beta  = (const float*)d_in[9];
  const float* w_end = (const float*)d_in[10];
  const float* gamma = (const float*)d_in[11];
  const float* bn_b  = (const float*)d_in[12];
  float* y = (float*)d_out;           // conv result lives in d_out

  char* ws = (char*)d_ws;
  const long MB = 1024 * 1024;
  float* sa = (float*)ws;                              // 32 MB
  float* ca = (float*)(ws + 32 * MB);                  // 32 MB
  char*  T  = ws + 64 * MB;                            // phase-aliased scratch

  // phase 1 (spatial attention)
  unsigned short* Xb = (unsigned short*)(T);                // 16 MB
  float* fb = (float*)(T + 16 * MB);                        // 4 MB
  float* fc = (float*)(T + 20 * MB);                        // 4 MB
  float* fd = (float*)(T + 24 * MB);                        // 32 MB
  unsigned short* Qp = (unsigned short*)(T + 56 * MB);      // 2 MB
  unsigned short* Kp = (unsigned short*)(T + 58 * MB);      // 2 MB
  unsigned short* Vp = (unsigned short*)(T + 60 * MB);      // 32 MB
  unsigned short* WbA = (unsigned short*)(T + 92 * MB);
  unsigned short* WcA = WbA + 8192;
  unsigned short* WdA = WcA + 8192;                         // +128 KB
  // phase 2 (channel attention)
  unsigned short* faA  = (unsigned short*)(T);              // 16 MB
  unsigned short* faB  = (unsigned short*)(T + 16 * MB);    // 16 MB
  float* attR = (float*)(T + 32 * MB);                      // 2 MB
  float* attS = (float*)(T + 34 * MB);                      // 2 MB
  unsigned short* attP = (unsigned short*)(T + 36 * MB);    // 1 MB
  unsigned short* faB2 = (unsigned short*)(T + 37 * MB);    // 16 MB
  // phase 3 (conv)
  float* xsum = (float*)(T);                                // 32 MB
  unsigned short* WA = (unsigned short*)(T + 32 * MB);      // ~1.2 MB
  float* bnscale = (float*)(T + 34 * MB);
  float* bnshift = bnscale + 256;

  // ---- phase 1: projections -------------------------------------------
  // Xb: per-batch B-frags of x_low (K=256,N=4096)
  pack_b_kernel<<<dim3(4096, 8), 256, 0, stream>>>(x_low, Xb, 256, 8, 4096, 1,
                                                   1048576, 1048576);
  pack_a_kernel<<<dim3(32, 1), 256, 0, stream>>>(wb, WbA, 2, 8, 256, 1, 0, 0);
  pack_a_kernel<<<dim3(32, 1), 256, 0, stream>>>(wc, WcA, 2, 8, 256, 1, 0, 0);
  pack_a_kernel<<<dim3(256, 1), 256, 0, stream>>>(wd, WdA, 16, 8, 256, 1, 0, 0);
  // fb/fc: [32][32768] (column index = b*4096+n); fd: [256][32768]
  gemm_lds_kernel<8><<<dim3(32, 2, 8), 128, 0, stream>>>(WbA, Xb, fb, 256, 32768,
      bb, nullptr, nullptr, 0, 1048576, 4096);
  gemm_lds_kernel<8><<<dim3(32, 2, 8), 128, 0, stream>>>(WcA, Xb, fc, 256, 32768,
      bc, nullptr, nullptr, 0, 1048576, 4096);
  gemm_lds_kernel<8><<<dim3(32, 16, 8), 128, 0, stream>>>(WdA, Xb, fd, 256, 32768,
      bd, nullptr, nullptr, 0, 1048576, 4096);
  // attention operand packs: Q (A-frags of fb^T), K (B-frags of fc), V (B-frags of fd^T)
  pack_a_kernel<<<dim3(4096, 1), 256, 0, stream>>>(fb, Qp, 2048, 1, 1, 32768, 0, 0);
  pack_b_kernel<<<dim3(4096, 1), 256, 0, stream>>>(fc, Kp, 2048, 1, 32768, 1, 0, 0);
  pack_b_kernel<<<dim3(65536, 1), 256, 0, stream>>>(fd, Vp, 16, 2048, 1, 32768, 0, 0);
  attn_kernel<<<dim3(256, 8), 32, 0, stream>>>(Qp, Kp, Vp, x_low, alpha, sa);

  // ---- phase 2: channel attention -------------------------------------
  pack_a_kernel<<<dim3(4096, 8), 256, 0, stream>>>(x_high, faA, 16, 128, 4096, 1,
                                                   1048576, 1048576);
  pack_b_kernel<<<dim3(4096, 8), 256, 0, stream>>>(x_high, faB, 16, 128, 1, 4096,
                                                   1048576, 1048576);
  gemm_bf16_kernel<<<dim3(4, 16, 8), 128, 0, stream>>>(faA, faB, attR, 16, 128, 256,
      nullptr, nullptr, nullptr, 1048576, 1048576, 65536);
  ca_softmax_kernel<<<2048, 256, 0, stream>>>(attR, attS);
  pack_a_kernel<<<dim3(256, 8), 256, 0, stream>>>(attS, attP, 16, 8, 256, 1,
                                                  65536, 65536);
  pack_b_kernel<<<dim3(4096, 8), 256, 0, stream>>>(x_high, faB2, 256, 8, 4096, 1,
                                                   1048576, 1048576);
  // ca = beta * (attS @ fa) + x_high
  gemm_lds_kernel<8><<<dim3(32, 16, 8), 128, 0, stream>>>(attP, faB2, ca, 256, 4096,
      nullptr, beta, x_high, 65536, 1048576, 1048576);

  // ---- phase 3: conv + BN + ReLU + fuse --------------------------------
  add_kernel<<<32768, 256, 0, stream>>>(x_low, x_high, xsum);
  pack_a_kernel<<<dim3(2304, 1), 256, 0, stream>>>(w_end, WA, 16, 72, 2304, 1, 0, 0);
  conv_kernel<<<dim3(2048, 4), 128, 0, stream>>>(WA, xsum, y);
  bn_stats_kernel<<<256, 256, 0, stream>>>(y, gamma, bn_b, bnscale, bnshift);
  final_kernel<<<32768, 256, 0, stream>>>(y, sa, ca, bnscale, bnshift);
}
